// MultiHeadDoublyStochasticSelfAttention_62612033241695
// MI455X (gfx1250) — matrix operand order repaired and checked by reading, not compiled, not run on hardware
//
#include <hip/hip_runtime.h>

typedef __bf16 bf16;
typedef __attribute__((ext_vector_type(16))) __bf16 v16bf;
typedef __attribute__((ext_vector_type(8)))  __bf16 v8bf;
typedef __attribute__((ext_vector_type(8)))  float  v8f;

#define FDIM 768
#define NSEQ 1024
#define HEADS 12
#define DHEAD 64
#define Q_ELEMS 6291456  // 8*12*1024*64

__device__ __forceinline__ v8f wmma_bf16(v16bf a, v16bf b, v8f c) {
  return __builtin_amdgcn_wmma_f32_16x16x32_bf16(false, a, false, b, (short)0, c, false, false);
}

// Build a 16-element bf16 fragment from two contiguous 8-element (16B) chunks.
__device__ __forceinline__ v16bf ld_frag16(const bf16* lo, const bf16* hi) {
  v16bf r;
  *(v8bf*)&r       = *(const v8bf*)lo;
  *((v8bf*)&r + 1) = *(const v8bf*)hi;
  return r;
}

// ---------------------------------------------------------------------------
// K1: fused QKV projection.  y = x @ W.T
//   Q,K written bf16 [b][h][n][d] (Q pre-scaled by d^-0.5)
//   V written bf16 TRANSPOSED [b][h][d][n] so attn@V B-frags are contiguous.
// grid = (8192/64, 768/64, 3), block = 256 (8 waves), WG tile 64x64.
// ---------------------------------------------------------------------------
__global__ __launch_bounds__(256) void qkv_proj_kernel(
    const float* __restrict__ x, const float* __restrict__ Wq,
    const float* __restrict__ Wk, const float* __restrict__ Wv,
    bf16* __restrict__ qb, bf16* __restrict__ kb, bf16* __restrict__ vb)
{
  __shared__ __align__(32) bf16 sA[64][32];
  __shared__ __align__(32) bf16 sB[64][32];   // B tile stored [col][k]
  const int m0 = blockIdx.x * 64;
  const int n0 = blockIdx.y * 64;
  const int z  = blockIdx.z;
  const float* W = (z == 0) ? Wq : (z == 1) ? Wk : Wv;
  bf16* out      = (z == 0) ? qb : (z == 1) ? kb : vb;
  const float scale = (z == 0) ? 0.125f : 1.0f;   // d^-0.5 = 1/8 folded into Q

  const int tid  = threadIdx.x;
  const int lane = tid & 31;
  const int w    = tid >> 5;
  const int tr   = (w & 3) * 16;
  const int tc   = (w >> 2) * 32;

  v8f acc0 = {}; v8f acc1 = {};

  for (int k0 = 0; k0 < FDIM; k0 += 32) {
    __syncthreads();
    {
      const int idx = tid * 8;
      const int mm = idx >> 5, kk = idx & 31;
      const float* src  = x + (size_t)(m0 + mm) * FDIM + k0 + kk;
      const float* srcb = W + (size_t)(n0 + mm) * FDIM + k0 + kk;  // W.T col tile
      #pragma unroll
      for (int e = 0; e < 8; ++e) sA[mm][kk + e] = (bf16)src[e];
      #pragma unroll
      for (int e = 0; e < 8; ++e) sB[mm][kk + e] = (bf16)srcb[e];
    }
    __syncthreads();
    const int arow = tr + (lane & 15);
    const int akb  = (lane < 16) ? 0 : 16;
    v16bf af = ld_frag16(&sA[arow][akb], &sA[arow][akb + 8]);
    const int bkb = (lane < 16) ? 0 : 16;
    v16bf b0 = *(const v16bf*)&sB[tc +      (lane & 15)][bkb];
    v16bf b1 = *(const v16bf*)&sB[tc + 16 + (lane & 15)][bkb];
    acc0 = wmma_bf16(af, b0, acc0);
    acc1 = wmma_bf16(af, b1, acc1);
  }

  #pragma unroll
  for (int half = 0; half < 2; ++half) {
    v8f a = half ? acc1 : acc0;
    const int N  = n0 + tc + half * 16 + (lane & 15);
    const int hd = N >> 6, dd = N & 63;
    #pragma unroll
    for (int r = 0; r < 8; ++r) {
      const int M  = m0 + tr + r + ((lane < 16) ? 0 : 8);
      const int bb = M >> 10, nr = M & 1023;
      size_t addr;
      if (z == 2)   // V transposed: [b][h][d][n]
        addr = ((size_t)(bb * HEADS + hd) * DHEAD + dd) * NSEQ + nr;
      else          // Q,K: [b][h][n][d]
        addr = ((size_t)(bb * HEADS + hd) * NSEQ + nr) * DHEAD + dd;
      out[addr] = (bf16)(a[r] * scale);
    }
  }
}

// ---------------------------------------------------------------------------
// K2: one sinkhorn half-iteration:
//   outv[r] = clog - logsumexp_c( (A[r,:].B[c,:]) + (use_add ? addv[c] : 0) )
// A,B are [1024][64] bf16 per (b,h).  grid=(64, 96), block=256.
// Row-pass: A=Q,B=K.  Col-pass: A=K,B=Q (dots^T), addv indexed by reduced axis.
// ---------------------------------------------------------------------------
__global__ __launch_bounds__(256) void sinkhorn_pass_kernel(
    const bf16* __restrict__ Amat, const bf16* __restrict__ Bmat,
    const float* __restrict__ addv, int use_add,
    float* __restrict__ outv, float clog)
{
  __shared__ float redM[8][16];
  __shared__ float redS[8][16];
  const int i0  = blockIdx.x * 16;
  const int bh  = blockIdx.y;
  const int tid = threadIdx.x, lane = tid & 31, w = tid >> 5;

  const bf16* A = Amat + (size_t)bh * NSEQ * DHEAD;
  const bf16* B = Bmat + (size_t)bh * NSEQ * DHEAD;
  const float* add = addv ? (addv + (size_t)bh * NSEQ) : nullptr;

  // A fragments for the 16 fixed rows, full K=64 (two K=32 frags).
  const int arow = i0 + (lane & 15);
  const int akb  = (lane < 16) ? 0 : 16;
  const bf16* arp = A + (size_t)arow * DHEAD;
  v16bf a0 = ld_frag16(arp + akb,      arp + akb + 8);
  v16bf a1 = ld_frag16(arp + 32 + akb, arp + 32 + akb + 8);

  float mx[8], ss[8];
  #pragma unroll
  for (int r = 0; r < 8; ++r) { mx[r] = -1e30f; ss[r] = 0.0f; }

  const int bkb = (lane < 16) ? 0 : 16;
  for (int it = 0; it < 4; ++it) {
    const int jb = it * 256 + w * 32;
    #pragma unroll
    for (int half = 0; half < 2; ++half) {
      const int jc = jb + half * 16 + (lane & 15);
      const bf16* bp = B + (size_t)jc * DHEAD;
      v16bf b0 = *(const v16bf*)(bp + bkb);
      v16bf b1 = *(const v16bf*)(bp + 32 + bkb);
      v8f c = {};
      c = wmma_bf16(a0, b0, c);
      c = wmma_bf16(a1, b1, c);
      const float av = use_add ? add[jc] : 0.0f;
      #pragma unroll
      for (int r = 0; r < 8; ++r) {          // streaming max-stable lse
        const float xv = c[r] + av;
        const float nm = fmaxf(mx[r], xv);
        ss[r] = ss[r] * __expf(mx[r] - nm) + __expf(xv - nm);
        mx[r] = nm;
      }
    }
  }
  // reduce across the 16 lanes of each half (rows r in lanes 0-15, r+8 in 16-31)
  #pragma unroll
  for (int r = 0; r < 8; ++r) {
    #pragma unroll
    for (int msk = 1; msk < 16; msk <<= 1) {
      const float om = __shfl_xor(mx[r], msk, 32);
      const float os = __shfl_xor(ss[r], msk, 32);
      const float nm = fmaxf(mx[r], om);
      ss[r] = ss[r] * __expf(mx[r] - nm) + os * __expf(om - nm);
      mx[r] = nm;
    }
  }
  if (lane == 0 || lane == 16) {
    const int rb = (lane == 0) ? 0 : 8;
    #pragma unroll
    for (int r = 0; r < 8; ++r) { redM[w][rb + r] = mx[r]; redS[w][rb + r] = ss[r]; }
  }
  __syncthreads();
  if (tid < 16) {
    float m = redM[0][tid], s = redS[0][tid];
    #pragma unroll
    for (int ww = 1; ww < 8; ++ww) {
      const float om = redM[ww][tid], os = redS[ww][tid];
      const float nm = fmaxf(m, om);
      s = s * __expf(m - nm) + os * __expf(om - nm);
      m = nm;
    }
    outv[(size_t)bh * NSEQ + i0 + tid] = clog - (m + __logf(s));
  }
}

// ---------------------------------------------------------------------------
// K3: out[i,:] = sum_j exp(q_i.k_j + u2_i + v1_j) * v[j,:]  ->  bf16 [b][n][768]
// V is stored transposed [b][h][d][n], so attn@V B-frags load straight from
// global memory (32B contiguous).  grid=(64, 96), block=256, dyn LDS = 32KB.
// ---------------------------------------------------------------------------
__global__ __launch_bounds__(256) void attn_v_kernel(
    const bf16* __restrict__ qb, const bf16* __restrict__ kb,
    const bf16* __restrict__ vtb, const float* __restrict__ u2,
    const float* __restrict__ v1, bf16* __restrict__ aout)
{
  extern __shared__ char smem[];
  bf16 (*sATT)[16][32] = (bf16(*)[16][32])smem;  // exp tile per wave (8KB)
  float* red = (float*)smem;                      // reuse after loop: [8][16][64]

  const int i0  = blockIdx.x * 16;
  const int bh  = blockIdx.y;
  const int bb  = bh / HEADS, hh = bh % HEADS;
  const int tid = threadIdx.x, lane = tid & 31, w = tid >> 5;

  const bf16* Q  = qb  + (size_t)bh * NSEQ * DHEAD;
  const bf16* K  = kb  + (size_t)bh * NSEQ * DHEAD;
  const bf16* Vt = vtb + (size_t)bh * DHEAD * NSEQ;   // [d][n]
  const float* u2p = u2 + (size_t)bh * NSEQ;
  const float* v1p = v1 + (size_t)bh * NSEQ;

  const int arow = i0 + (lane & 15);
  const int akb  = (lane < 16) ? 0 : 16;
  const bf16* arp = Q + (size_t)arow * DHEAD;
  v16bf a0 = ld_frag16(arp + akb,      arp + akb + 8);
  v16bf a1 = ld_frag16(arp + 32 + akb, arp + 32 + akb + 8);

  float u2row[8];
  #pragma unroll
  for (int r = 0; r < 8; ++r) u2row[r] = u2p[i0 + r + ((lane < 16) ? 0 : 8)];

  v8f acc[4] = {{}, {}, {}, {}};
  const int bkb = (lane < 16) ? 0 : 16;

  for (int it = 0; it < 4; ++it) {
    const int jb = it * 256 + w * 32;
    // dots + bias + exp -> wave-private LDS tile (A-layout repack)
    #pragma unroll
    for (int half = 0; half < 2; ++half) {
      const int jc = jb + half * 16 + (lane & 15);
      const bf16* bp = K + (size_t)jc * DHEAD;
      v16bf b0 = *(const v16bf*)(bp + bkb);
      v16bf b1 = *(const v16bf*)(bp + 32 + bkb);
      v8f c = {};
      c = wmma_bf16(a0, b0, c);
      c = wmma_bf16(a1, b1, c);
      const float vj = v1p[jc];
      const int jl = half * 16 + (lane & 15);
      const int mb = (lane < 16) ? 0 : 8;
      #pragma unroll
      for (int r = 0; r < 8; ++r)
        sATT[w][mb + r][jl] = (bf16)__expf(c[r] + u2row[r] + vj);
    }
    __syncthreads();
    {
      const int irow = lane & 15;
      v16bf aa = ld_frag16(&sATT[w][irow][akb], &sATT[w][irow][akb + 8]);
      #pragma unroll
      for (int t = 0; t < 4; ++t) {   // V^T B-frags straight from global
        const int dd = t * 16 + (lane & 15);
        v16bf bt = *(const v16bf*)(Vt + (size_t)dd * NSEQ + jb + bkb);
        acc[t] = wmma_bf16(aa, bt, acc[t]);
      }
    }
    __syncthreads();
  }
  // cross-wave reduction of the 16x64 partial tiles (red aliases sATT; last
  // __syncthreads above guarantees all sATT reads completed)
  #pragma unroll
  for (int t = 0; t < 4; ++t) {
    const int dd = t * 16 + (lane & 15);
    const int mb = (lane < 16) ? 0 : 8;
    #pragma unroll
    for (int r = 0; r < 8; ++r) red[(w * 16 + mb + r) * 64 + dd] = acc[t][r];
  }
  __syncthreads();
  {
    const int idx = tid * 4;
    const int m = idx >> 6, dd0 = idx & 63;
    #pragma unroll
    for (int e = 0; e < 4; ++e) {
      float s = 0.0f;
      #pragma unroll
      for (int ww = 0; ww < 8; ++ww) s += red[(ww * 16 + m) * 64 + dd0 + e];
      aout[((size_t)(bb * NSEQ + i0 + m)) * FDIM + hh * DHEAD + dd0 + e] = (bf16)s;
    }
  }
}

// ---------------------------------------------------------------------------
// K4: final projection  out = attn_out @ Wo.T + bo   (f32 output)
// A-tiles are raw bf16 -> moved with global_load_async_to_lds_b128 (ASYNCcnt).
// ---------------------------------------------------------------------------
__global__ __launch_bounds__(256) void out_proj_kernel(
    const bf16* __restrict__ aout, const float* __restrict__ Wo,
    const float* __restrict__ bo, float* __restrict__ out)
{
  __shared__ __align__(32) bf16 sA[64][32];
  __shared__ __align__(32) bf16 sB[64][32];
  const int m0 = blockIdx.x * 64, n0 = blockIdx.y * 64;
  const int tid = threadIdx.x, lane = tid & 31, w = tid >> 5;
  const int tr = (w & 3) * 16, tc = (w >> 2) * 32;
  v8f acc0 = {}, acc1 = {};

  // per-thread A async-copy addressing: 16B per thread covers the 4KB tile
  const int amm  = tid >> 2;             // row 0..63
  const int asub = (tid & 3) * 8;        // bf16 element offset in row
  const unsigned lds_a =
      (unsigned)(unsigned long long)(__attribute__((address_space(3))) bf16*)&sA[amm][asub];

  for (int k0 = 0; k0 < FDIM; k0 += 32) {
    __syncthreads();
    {
      // A tile: raw bf16 bytes, async DMA into LDS (no VGPR round trip)
      const bf16* gsrc = aout + (size_t)(m0 + amm) * FDIM + k0 + asub;
      asm volatile("global_load_async_to_lds_b128 %0, %1, off"
                   :: "v"(lds_a), "v"(gsrc) : "memory");
      // B tile: f32 -> bf16 convert, regular DS stores
      const int idx = tid * 8;
      const int mm = idx >> 5, kk = idx & 31;
      const float* srcb = Wo + (size_t)(n0 + mm) * FDIM + k0 + kk;
      #pragma unroll
      for (int e = 0; e < 8; ++e) sB[mm][kk + e] = (bf16)srcb[e];
    }
    asm volatile("s_wait_asynccnt 0" ::: "memory");
    __syncthreads();
    const int arow = tr + (lane & 15);
    const int akb  = (lane < 16) ? 0 : 16;
    v16bf af = ld_frag16(&sA[arow][akb], &sA[arow][akb + 8]);
    const int bkb = (lane < 16) ? 0 : 16;
    v16bf b0 = *(const v16bf*)&sB[tc +      (lane & 15)][bkb];
    v16bf b1 = *(const v16bf*)&sB[tc + 16 + (lane & 15)][bkb];
    acc0 = wmma_bf16(af, b0, acc0);
    acc1 = wmma_bf16(af, b1, acc1);
  }
  #pragma unroll
  for (int half = 0; half < 2; ++half) {
    v8f a = half ? acc1 : acc0;
    const int N = n0 + tc + half * 16 + (lane & 15);
    const float bias = bo[N];
    #pragma unroll
    for (int r = 0; r < 8; ++r) {
      const int M = m0 + tr + r + ((lane < 16) ? 0 : 8);
      out[(size_t)M * FDIM + N] = a[r] + bias;
    }
  }
}

// ---------------------------------------------------------------------------
extern "C" void kernel_launch(void* const* d_in, const int* in_sizes, int n_in,
                              void* d_out, int out_size, void* d_ws, size_t ws_size,
                              hipStream_t stream) {
  const float* x  = (const float*)d_in[0];
  const float* Wq = (const float*)d_in[1];
  const float* Wk = (const float*)d_in[2];
  const float* Wv = (const float*)d_in[3];
  const float* Wo = (const float*)d_in[4];
  const float* bo = (const float*)d_in[5];
  float* out = (float*)d_out;

  char* ws = (char*)d_ws;
  const size_t QB = (size_t)Q_ELEMS * 2;      // 12.58 MB per bf16 tensor
  bf16*  qb = (bf16*)(ws);
  bf16*  kb = (bf16*)(ws + QB);
  bf16*  vb = (bf16*)(ws + 2 * QB);           // transposed [b][h][d][n]
  bf16*  ao = (bf16*)(ws + 3 * QB);           // attn_out bf16 [b][n][768]
  float* u1 = (float*)(ws + 4 * QB);
  float* v1 = (float*)(ws + 4 * QB + 393216);
  float* u2 = (float*)(ws + 4 * QB + 2 * 393216);

  const float log_mu = -logf(1024.0f);        // log(1/n) == log(1/m)

  dim3 g1(128, 12, 3);
  qkv_proj_kernel<<<g1, 256, 0, stream>>>(x, Wq, Wk, Wv, qb, kb, vb);

  dim3 g2(64, 96);
  // u1 = log_mu - lse_j(dots)
  sinkhorn_pass_kernel<<<g2, 256, 0, stream>>>(qb, kb, nullptr, 0, u1, log_mu);
  // v1 = log_nu - lse_i(dots + u1_i)   (transposed: A=K, B=Q)
  sinkhorn_pass_kernel<<<g2, 256, 0, stream>>>(kb, qb, u1, 1, v1, log_mu);
  // u2 = log_mu - lse_j(dots + v1_j)
  sinkhorn_pass_kernel<<<g2, 256, 0, stream>>>(qb, kb, v1, 1, u2, log_mu);

  attn_v_kernel<<<g2, 256, 32768, stream>>>(qb, kb, vb, u2, v1, ao);

  dim3 g4(128, 12);
  out_proj_kernel<<<g4, 256, 0, stream>>>(ao, Wo, bo, out);
}
